// EGBRNN_42649025249294
// MI455X (gfx1250) — matrix operand order repaired and checked by reading, not compiled, not run on hardware
//
#include <hip/hip_runtime.h>
#include <math.h>

// ---------------------------------------------------------------------------
// EGBRNN (learned Kalman RNN) for MI455X / gfx1250.
// B=2048, T=128, H=256. 128 workgroups x 128 threads (4 wave32).
// Each WG owns a 16-row batch tile; waves split N=256 into 4x64 columns.
// bf16 WMMA 16x16x32, f32 accumulate. Weights pre-transposed to [N][Kpad] bf16
// and kept L2-resident; B fragments are re-loaded from global every step.
// An opaque *integer offset* defeats LICM (no hoist -> no scratch spills)
// while keeping kernel-arg pointer provenance so loads stay global_load,
// not flat_load (flat would also burn DScnt and 64-bit VGPR address math).
// ---------------------------------------------------------------------------

typedef __bf16 bf16;
typedef __attribute__((ext_vector_type(16))) __bf16 v16bf;
typedef __attribute__((ext_vector_type(8)))  __bf16 v8bf;
typedef __attribute__((ext_vector_type(8)))  float  v8f;

#define APAD   288   // padded K for first-layer matmuls (max K=262 -> 9 x 32)
#define HSTR   256   // hidden activation stride
#define TSTEPS 128
#define BTILE  16

// Unknown-but-zero element offset, regenerated every loop iteration: weight
// addresses become loop-variant so B-fragment loads cannot be hoisted+spilled.
__device__ __forceinline__ long opaque_zero() {
  long z = 0;
  asm volatile("" : "+s"(z));
  return z;
}

// Tell InferAddressSpaces this pointer is global (not LDS/scratch) so the
// backend selects global_load (LOADcnt only) instead of flat_load (+DScnt).
__device__ __forceinline__ const bf16* ghint(const bf16* p) {
  __builtin_assume(!__builtin_amdgcn_is_shared((const void*)p) &&
                   !__builtin_amdgcn_is_private((const void*)p));
  return p;
}

// --- A fragment (16x32 bf16, row per lane): ISA 7.12.2 layout ---------------
// lanes 0-15 : row M=lane,   K = k0+0..7   and k0+16..23
// lanes 16-31: row M=lane-16,K = k0+8..15  and k0+24..31
__device__ __forceinline__ v16bf load_a_frag(const bf16* A, int stride, int k0, int lane) {
  const int row  = lane & 15;
  const int koff = k0 + ((lane & 16) ? 8 : 0);
  const bf16* p  = A + row * stride + koff;
  v8bf lo = *(const v8bf*)(p);
  v8bf hi = *(const v8bf*)(p + 16);
  v16bf r;
#pragma unroll
  for (int i = 0; i < 8; ++i) { r[i] = lo[i]; r[i + 8] = hi[i]; }
  return r;
}

// --- B fragment (32x16 bf16, column per lane) from pre-transposed W^T -------
// lanes 0-15 : col N=lane,    K = k0+0..7   and k0+16..23
// lanes 16-31: col N=lane-16, K = k0+8..15  and k0+24..31
__device__ __forceinline__ v16bf load_b_frag(const bf16* Wt, int kpad, int ntile, int k0, int lane) {
  const int n    = ntile * 16 + (lane & 15);
  const int koff = k0 + ((lane & 16) ? 8 : 0);
  const bf16* p  = Wt + (size_t)n * kpad + koff;
  v8bf lo = *(const v8bf*)(p);
  v8bf hi = *(const v8bf*)(p + 16);
  v16bf r;
#pragma unroll
  for (int i = 0; i < 8; ++i) { r[i] = lo[i]; r[i + 8] = hi[i]; }
  return r;
}

__device__ __forceinline__ v8f wmma_bf16(v16bf a, v16bf b, v8f c) {
  return __builtin_amdgcn_wmma_f32_16x16x32_bf16(false, a, false, b, (short)0, c, false, false);
}

// D[16x256] = tanh(A[16xKpad] @ W[Kx256] + bias). Each wave computes 64 cols.
// If inplace, barrier after all A reads before overwriting A with the result.
__device__ __forceinline__ void big_mm(const bf16* Alds, const bf16* Wt,
                                       const float* __restrict__ bias,
                                       bf16* Dlds, int dstride, int lane, int wv, bool inplace) {
  Wt = ghint(Wt);
  v8f z = {};
  v8f acc[4] = {z, z, z, z};
#pragma unroll
  for (int ks = 0; ks < 9; ++ks) {
    v16bf a = load_a_frag(Alds, APAD, ks * 32, lane);
#pragma unroll
    for (int nt = 0; nt < 4; ++nt) {
      v16bf b = load_b_frag(Wt, APAD, wv * 4 + nt, ks * 32, lane);
      acc[nt] = wmma_bf16(a, b, acc[nt]);
    }
  }
  if (inplace) __syncthreads();
  const int ncol  = lane & 15;           // C/D layout: N = lane%16
  const int rbase = (lane & 16) ? 8 : 0; // M = vgpr + (lane>=16 ? 8 : 0)
#pragma unroll
  for (int nt = 0; nt < 4; ++nt) {
    const int col  = wv * 64 + nt * 16 + ncol;
    const float bb = bias[col];
#pragma unroll
    for (int i = 0; i < 8; ++i)
      Dlds[(rbase + i) * dstride + col] = (bf16)tanhf(acc[nt][i] + bb);
  }
}

// out[16xN] (N<=16) = h[16x256] @ W2[256xN] + b2 ; W2 pre-transposed to [16][256]
__device__ __forceinline__ void small_mm(const bf16* h, const bf16* W2t,
                                         const float* __restrict__ b2,
                                         int ncols, float* outArr, int lane) {
  W2t = ghint(W2t);
  v8f acc = {};
#pragma unroll
  for (int ks = 0; ks < 8; ++ks) {
    v16bf a = load_a_frag(h, HSTR, ks * 32, lane);
    v16bf b = load_b_frag(W2t, HSTR, 0, ks * 32, lane);
    acc = wmma_bf16(a, b, acc);
  }
  const int n     = lane & 15;
  const int rbase = (lane & 16) ? 8 : 0;
  if (n < ncols) {
#pragma unroll
    for (int i = 0; i < 8; ++i)
      outArr[(rbase + i) * ncols + n] = acc[i] + b2[n];
  }
}

// --- weight converter: W[KxN] f32 row-major -> W^T[Npad x Kpad] bf16, zero pad
__global__ void conv_weight(const float* __restrict__ W, int K, int N,
                            bf16* __restrict__ outT, int Kpad, int Npad) {
  int idx = blockIdx.x * blockDim.x + threadIdx.x;
  if (idx >= Kpad * Npad) return;
  int n = idx / Kpad;
  int k = idx - n * Kpad;
  float v = (n < N && k < K) ? W[(size_t)k * N + n] : 0.0f;
  outT[(size_t)n * Kpad + k] = (bf16)v;
}

// ---------------------------------------------------------------------------
__global__ __launch_bounds__(128) void egbrnn_kernel(
    const float* __restrict__ x, const float* __restrict__ target, const float* __restrict__ c0,
    const bf16* __restrict__ Wm0t,  const float* __restrict__ bm0,
    const bf16* __restrict__ Wm1t,  const float* __restrict__ bm1,
    const bf16* __restrict__ Wf10t, const float* __restrict__ bf10,
    const bf16* __restrict__ Wf11t, const float* __restrict__ bf11,
    const bf16* __restrict__ Wf20t, const float* __restrict__ bf20,
    const bf16* __restrict__ Wf21t, const float* __restrict__ bf21,
    const bf16* __restrict__ Wh0t,  const float* __restrict__ bh0,
    const bf16* __restrict__ Wh1t,  const float* __restrict__ bh1,
    float* __restrict__ out)
{
  __shared__ __align__(16) bf16 Abuf[16 * APAD]; // [c|cu , tail(6) , zero pad]
  __shared__ __align__(16) bf16 H1[16 * HSTR];
  __shared__ __align__(16) bf16 H2[16 * HSTR];
  __shared__ __align__(16) bf16 H3[16 * HSTR];
  __shared__ float dArr[16 * 2];
  __shared__ float pvArr[16 * 4];
  __shared__ float hvArr[16 * 2];

  const int tid  = threadIdx.x;
  const int lane = tid & 31;
  const int wv   = tid >> 5;
  const int b0   = blockIdx.x * BTILE;

  // init: c0 -> Abuf cols 0..255 ; zero cols 256..287
  for (int i = tid; i < 16 * 256; i += 128) {
    int row = i >> 8, col = i & 255;
    Abuf[row * APAD + col] = (bf16)c0[(size_t)(b0 + row) * 256 + col];
  }
  for (int i = tid; i < 16 * 32; i += 128) {
    int row = i >> 5, col = 256 + (i & 31);
    Abuf[row * APAD + col] = (bf16)0.0f;
  }

  // Kalman state: one batch row per lane (tid<16), kept in registers for all T
  float s0 = 0.f, s1 = 0.f, s2 = 0.f, s3 = 0.f;
  float P[16];
  if (tid < 16) {
    const size_t b = (size_t)(b0 + tid);
    s0 = target[(b * TSTEPS + 0) * 4 + 0];
    s1 = target[(b * TSTEPS + 0) * 4 + 1];
    s2 = target[(b * TSTEPS + 0) * 4 + 2];
    s3 = target[(b * TSTEPS + 0) * 4 + 3];
#pragma unroll
    for (int i = 0; i < 16; ++i) P[i] = ((i >> 2) == (i & 3)) ? 1.0f : 0.0f;
  }
  __syncthreads();

  float m0f = 0.f, m1f = 0.f;
  for (int t = 0; t < TSTEPS; ++t) {
    // Loop-variant zero offset: weight addresses can't be hoisted out of the
    // t-loop, but pointer provenance (kernel arg -> global) is preserved.
    const long zo = opaque_zero();
    const bf16* wm0  = Wm0t  + zo;
    const bf16* wm1  = Wm1t  + zo;
    const bf16* wf10 = Wf10t + zo;
    const bf16* wf11 = Wf11t + zo;
    const bf16* wf20 = Wf20t + zo;
    const bf16* wf21 = Wf21t + zo;
    const bf16* wh0  = Wh0t  + zo;
    const bf16* wh1  = Wh1t  + zo;

    // ---- build A tail = [sn(4), mn(2)] for this step ----
    if (tid < 16) {
      const size_t b = (size_t)(b0 + tid);
      m0f = x[(b * TSTEPS + t) * 2 + 0];
      m1f = x[(b * TSTEPS + t) * 2 + 1];
      bf16* ar = Abuf + tid * APAD;
      ar[256] = (bf16)(s0 * (1.f / 60.f));
      ar[257] = (bf16)(s1 * (1.f / 60.f));
      ar[258] = (bf16)(s2 * (1.f / 60.f));
      ar[259] = (bf16)(s3 * (1.f / 60.f));
      ar[260] = (bf16)(m0f * (1.f / 60.f));
      ar[261] = (bf16)(m1f * (1.f / 60.f));
    }
    __syncthreads();

    // cu = tanh([c,sn] @ Wm0 + bm0)  (in-place into Abuf cols 0..255)
    big_mm(Abuf, wm0, bm0, Abuf, APAD, lane, wv, true);
    __syncthreads();

    // h1 = tanh([cu,sn] @ Wf10 + bf10) ; h2 = tanh([cu,sn] @ Wf20 + bf20)
    // h3 = tanh([cu,sn,mn] @ Wh0 + bh0)   (zero-padded weight rows kill extras)
    big_mm(Abuf, wf10, bf10, H1, HSTR, lane, wv, false);
    big_mm(Abuf, wf20, bf20, H2, HSTR, lane, wv, false);
    big_mm(Abuf, wh0,  bh0,  H3, HSTR, lane, wv, false);
    __syncthreads();

    // tiny second layers, one wave each (wave-uniform branches: EXEC all-ones)
    if      (wv == 0) small_mm(H1, wf11, bf11, 2, dArr,  lane);
    else if (wv == 1) small_mm(H2, wf21, bf21, 4, pvArr, lane);
    else if (wv == 2) small_mm(H3, wh1,  bh1,  2, hvArr, lane);
    __syncthreads();

    // ---- per-row Kalman update (f32 VALU) ----
    if (tid < 16) {
      const float d0 = dArr[tid * 2 + 0], d1 = dArr[tid * 2 + 1];
      float pvv[4] = { pvArr[tid * 4 + 0], pvArr[tid * 4 + 1],
                       pvArr[tid * 4 + 2], pvArr[tid * 4 + 3] };
      const float hv0 = hvArr[tid * 2 + 0], hv1 = hvArr[tid * 2 + 1];

      float sp[4];
      sp[0] = s0 + s2 + 0.5f * d0;
      sp[1] = s1 + s3 + 0.5f * d1;
      sp[2] = s2 + d0;
      sp[3] = s3 + d1;

      float FP_[16], Pp[16];
#pragma unroll
      for (int j = 0; j < 4; ++j) {
        FP_[0 * 4 + j] = P[0 * 4 + j] + P[2 * 4 + j];
        FP_[1 * 4 + j] = P[1 * 4 + j] + P[3 * 4 + j];
        FP_[2 * 4 + j] = P[2 * 4 + j];
        FP_[3 * 4 + j] = P[3 * 4 + j];
      }
#pragma unroll
      for (int i = 0; i < 4; ++i) {
        Pp[i * 4 + 0] = FP_[i * 4 + 0] + FP_[i * 4 + 2];
        Pp[i * 4 + 1] = FP_[i * 4 + 1] + FP_[i * 4 + 3];
        Pp[i * 4 + 2] = FP_[i * 4 + 2];
        Pp[i * 4 + 3] = FP_[i * 4 + 3];
      }
#pragma unroll
      for (int i = 0; i < 4; ++i) {
#pragma unroll
        for (int j = 0; j < 4; ++j) Pp[i * 4 + j] += pvv[i] * pvv[j];
        Pp[i * 4 + i] += 0.01f;
      }
      const float in0 = m0f - sp[0];
      const float in1 = m1f - sp[1];
      const float S00 = Pp[0] + hv0 * hv0 + 1.f;
      const float S01 = Pp[1] + hv0 * hv1;
      const float S10 = Pp[4] + hv1 * hv0;
      const float S11 = Pp[5] + hv1 * hv1 + 1.f;
      const float inv = 1.f / (S00 * S11 - S01 * S10);
      const float i00 =  S11 * inv, i01 = -S01 * inv;
      const float i10 = -S10 * inv, i11 =  S00 * inv;
      float K_[8], KS[8], su[4];
#pragma unroll
      for (int i = 0; i < 4; ++i) {
        K_[i * 2 + 0] = Pp[i * 4 + 0] * i00 + Pp[i * 4 + 1] * i10;
        K_[i * 2 + 1] = Pp[i * 4 + 0] * i01 + Pp[i * 4 + 1] * i11;
        su[i] = sp[i] + K_[i * 2 + 0] * in0 + K_[i * 2 + 1] * in1;
        KS[i * 2 + 0] = K_[i * 2 + 0] * S00 + K_[i * 2 + 1] * S10;
        KS[i * 2 + 1] = K_[i * 2 + 0] * S01 + K_[i * 2 + 1] * S11;
      }
#pragma unroll
      for (int i = 0; i < 4; ++i)
#pragma unroll
        for (int j = 0; j < 4; ++j)
          P[i * 4 + j] = Pp[i * 4 + j] - (KS[i * 2 + 0] * K_[j * 2 + 0] + KS[i * 2 + 1] * K_[j * 2 + 1]);

      const size_t b = (size_t)(b0 + tid);
      float* op = out + (b * TSTEPS + t) * 4;
      op[0] = su[0]; op[1] = su[1]; op[2] = su[2]; op[3] = su[3];
      s0 = su[0]; s1 = su[1]; s2 = su[2]; s3 = su[3];

      // A tail for c_new: [cu | mn(2) | s_upd/60 (4)]
      bf16* ar = Abuf + tid * APAD;
      ar[256] = (bf16)(m0f  * (1.f / 60.f));
      ar[257] = (bf16)(m1f  * (1.f / 60.f));
      ar[258] = (bf16)(su[0] * (1.f / 60.f));
      ar[259] = (bf16)(su[1] * (1.f / 60.f));
      ar[260] = (bf16)(su[2] * (1.f / 60.f));
      ar[261] = (bf16)(su[3] * (1.f / 60.f));
    }
    __syncthreads();

    // c_new = tanh([cu,mn,sun] @ Wm1 + bm1)  (in-place -> carried state)
    big_mm(Abuf, wm1, bm1, Abuf, APAD, lane, wv, true);
    __syncthreads();
  }
}

// ---------------------------------------------------------------------------
extern "C" void kernel_launch(void* const* d_in, const int* in_sizes, int n_in,
                              void* d_out, int out_size, void* d_ws, size_t ws_size,
                              hipStream_t stream) {
  const float* x      = (const float*)d_in[0];
  const float* target = (const float*)d_in[1];
  const float* c0     = (const float*)d_in[2];
  const float* Wm0  = (const float*)d_in[3];  const float* bm0  = (const float*)d_in[4];
  const float* Wm1  = (const float*)d_in[5];  const float* bm1  = (const float*)d_in[6];
  const float* Wf10 = (const float*)d_in[7];  const float* bf10 = (const float*)d_in[8];
  const float* Wf11 = (const float*)d_in[9];  const float* bf11 = (const float*)d_in[10];
  const float* Wf20 = (const float*)d_in[11]; const float* bf20 = (const float*)d_in[12];
  const float* Wf21 = (const float*)d_in[13]; const float* bf21 = (const float*)d_in[14];
  const float* Wh0  = (const float*)d_in[15]; const float* bh0  = (const float*)d_in[16];
  const float* Wh1  = (const float*)d_in[17]; const float* bh1  = (const float*)d_in[18];

  const size_t BIG = (size_t)256 * APAD;  // 73728 bf16 elems per big weight
  const size_t SML = (size_t)16 * 256;    // 4096 per small weight
  bf16* ws    = (bf16*)d_ws;
  bf16* Wm0t  = ws;
  bf16* Wm1t  = Wm0t  + BIG;
  bf16* Wf10t = Wm1t  + BIG;
  bf16* Wf20t = Wf10t + BIG;
  bf16* Wh0t  = Wf20t + BIG;
  bf16* Wf11t = Wh0t  + BIG;
  bf16* Wf21t = Wf11t + SML;
  bf16* Wh1t  = Wf21t + SML;

  const int thr = 256;
  const int gbig = (int)((BIG + thr - 1) / thr);
  const int gsml = (int)((SML + thr - 1) / thr);
  conv_weight<<<gbig, thr, 0, stream>>>(Wm0,  260, 256, Wm0t,  APAD, 256);
  conv_weight<<<gbig, thr, 0, stream>>>(Wm1,  262, 256, Wm1t,  APAD, 256);
  conv_weight<<<gbig, thr, 0, stream>>>(Wf10, 260, 256, Wf10t, APAD, 256);
  conv_weight<<<gbig, thr, 0, stream>>>(Wf20, 260, 256, Wf20t, APAD, 256);
  conv_weight<<<gbig, thr, 0, stream>>>(Wh0,  262, 256, Wh0t,  APAD, 256);
  conv_weight<<<gsml, thr, 0, stream>>>(Wf11, 256, 2, Wf11t, 256, 16);
  conv_weight<<<gsml, thr, 0, stream>>>(Wf21, 256, 4, Wf21t, 256, 16);
  conv_weight<<<gsml, thr, 0, stream>>>(Wh1,  256, 2, Wh1t,  256, 16);

  egbrnn_kernel<<<2048 / BTILE, 128, 0, stream>>>(
      x, target, c0,
      Wm0t, bm0, Wm1t, bm1, Wf10t, bf10, Wf11t, bf11,
      Wf20t, bf20, Wf21t, bf21, Wh0t, bh0, Wh1t, bh1,
      (float*)d_out);
}
